// pw_Cheb_GCNN1_46755013984839
// MI455X (gfx1250) — matrix operand order repaired and checked by reading, not compiled, not run on hardware
//
#include <hip/hip_runtime.h>
#include <hip/hip_bf16.h>
#include <stddef.h>

#define N_NODES 8192
#define IN_F    2048
#define OUT_MF  1024
#define N_CONN  16384
#define E_EDGES 262144
#define CL1_F   64
#define OUT_F   2
#define EPSF    1e-5f

typedef __attribute__((ext_vector_type(16))) __bf16 bf16x16;
typedef __attribute__((ext_vector_type(8)))  float  floatx8;

// f32 -> bf16 with round-to-nearest-even (bit-level)
static __device__ __forceinline__ __bf16 f2bf(float f) {
    unsigned u = __builtin_bit_cast(unsigned, f);
    unsigned r = (u + 0x7FFFu + ((u >> 16) & 1u)) >> 16;
    unsigned short s = (unsigned short)r;
    return __builtin_bit_cast(__bf16, s);
}

// ---------------------------------------------------------------------------
// Kernel 1: SparseMaskedLinear + bias + ReLU + BN(eval).  One block per node.
// x row (8KB) and the 1024-wide accumulator (4KB) live in LDS; 16384
// gather/scatter MACs via ds_add_f32.
// ---------------------------------------------------------------------------
__global__ void k_sparse_linear(const float* __restrict__ x,
                                const int*   __restrict__ sparse_mask, // (N_CONN,2)
                                const float* __restrict__ w_sparse,
                                const float* __restrict__ b_sparse,
                                const float* __restrict__ g,
                                const float* __restrict__ bb,
                                const float* __restrict__ mean,
                                const float* __restrict__ var,
                                float* __restrict__ h)
{
    __shared__ float xrow[IN_F];
    __shared__ float acc[OUT_MF];
    const int node = blockIdx.x;
    const float* xr = x + (size_t)node * IN_F;
    for (int i = threadIdx.x; i < IN_F; i += blockDim.x) xrow[i] = xr[i];
    for (int i = threadIdx.x; i < OUT_MF; i += blockDim.x) acc[i] = 0.0f;
    __syncthreads();
    for (int c = threadIdx.x; c < N_CONN; c += blockDim.x) {
        int ci = sparse_mask[2 * c + 0];
        int co = sparse_mask[2 * c + 1];
        atomicAdd(&acc[co], xrow[ci] * w_sparse[c]); // ds_add_f32
    }
    __syncthreads();
    float* hr = h + (size_t)node * OUT_MF;
    for (int i = threadIdx.x; i < OUT_MF; i += blockDim.x) {
        float t = acc[i] + b_sparse[i];
        t = fmaxf(t, 0.0f);
        t = (t - mean[i]) * (g[i] * rsqrtf(var[i] + EPSF)) + bb[i];
        hr[i] = t;
    }
}

// ---------------------------------------------------------------------------
// Degree + edge normalization
// ---------------------------------------------------------------------------
__global__ void k_zero(float* __restrict__ p, size_t n) {
    size_t i = (size_t)blockIdx.x * blockDim.x + threadIdx.x;
    if (i < n) p[i] = 0.0f;
}

__global__ void k_deg(const int* __restrict__ ei, const float* __restrict__ ew,
                      float* __restrict__ deg) {
    int e = blockIdx.x * blockDim.x + threadIdx.x;
    if (e < E_EDGES) atomicAdd(&deg[ei[e]], ew[e]); // row = ei[0][e]
}

__global__ void k_normw(const int* __restrict__ ei, const float* __restrict__ ew,
                        const float* __restrict__ deg, float* __restrict__ nw) {
    int e = blockIdx.x * blockDim.x + threadIdx.x;
    if (e >= E_EDGES) return;
    int r = ei[e], c = ei[E_EDGES + e];
    float dr = deg[r], dc = deg[c];
    float ir = dr > 0.0f ? rsqrtf(fmaxf(dr, EPSF)) : 0.0f;
    float ic = dc > 0.0f ? rsqrtf(fmaxf(dc, EPSF)) : 0.0f;
    nw[e] = -(ir * ew[e] * ic);
}

// ---------------------------------------------------------------------------
// SpMM scatter: y[col[e], :] += nw[e] * z[row[e], :]
// One block (256 threads) per edge; each thread loads one float4 of the
// source row (global_load_b128) and issues 4 global_atomic_add_f32.
// ---------------------------------------------------------------------------
__global__ void k_spmm(const int* __restrict__ ei, const float* __restrict__ nw,
                       const float* __restrict__ z, float* __restrict__ y) {
    int e = blockIdx.x;
    int r = ei[e], c = ei[E_EDGES + e];
    float w = nw[e];
    const float4* zr = (const float4*)(z + (size_t)r * OUT_MF);
    float*        yc = y + (size_t)c * OUT_MF;
    float4 v = zr[threadIdx.x];            // 256 threads * 4 = 1024 features
    int f = threadIdx.x * 4;
    atomicAdd(&yc[f + 0], w * v.x);
    atomicAdd(&yc[f + 1], w * v.y);
    atomicAdd(&yc[f + 2], w * v.z);
    atomicAdd(&yc[f + 3], w * v.w);
}

// tx2 = 2*y - tx0  (in place over y)
__global__ void k_cheb_combine(float* __restrict__ y, const float* __restrict__ x0, size_t n) {
    size_t i = (size_t)blockIdx.x * blockDim.x + threadIdx.x;
    if (i < n) y[i] = 2.0f * y[i] - x0[i];
}

// ---------------------------------------------------------------------------
// WMMA GEMM:  C[N_NODES x 64] (+)= A[N_NODES x 1024] @ B[1024 x 64]
// bf16 inputs, f32 accumulate via v_wmma_f32_16x16x32_bf16.
// Block = 256 threads = 8 waves = (2 M-groups) x (4 N-tiles); each wave
// computes TWO 16x16 C tiles stacked in M that share one B fragment
// -> block covers 64 rows x 64 cols, two WMMA per k-step, 32 k-steps.
// Fragment packing per ISA 7.12.2:
//   A (16x32 bf16): lane m = L%16, half = L/16; V0..3 -> K = half*8 + 2v+s,
//                   V4..7 -> K = 16 + half*8 + 2(v-4)+s
//   B (32x16 bf16): lane n = L%16, half = L/16; same K mapping as A
//   C (16x16 f32):  lane n = L%16, Vr -> m = r + 8*(L/16)
// ---------------------------------------------------------------------------
__global__ void k_gemm_wmma(const float* __restrict__ A,
                            const float* __restrict__ B,
                            float* __restrict__ C,
                            int accumulate)
{
    const int lane = threadIdx.x & 31;
    const int wave = threadIdx.x >> 5;
    const int colBase = (wave & 3) * 16;
    const int mBase   = blockIdx.x * 64 + (wave >> 2) * 32;
    const int half = lane >> 4;
    const int lm   = lane & 15;
    const int n    = colBase + lm;
    const int crow0 = mBase + 8 * half;       // C rows for tile 0
    const int crow1 = mBase + 16 + 8 * half;  // C rows for tile 1

    floatx8 c0, c1;
    if (accumulate) {
#pragma unroll
        for (int r = 0; r < 8; ++r) {
            c0[r] = C[(size_t)(crow0 + r) * CL1_F + n];
            c1[r] = C[(size_t)(crow1 + r) * CL1_F + n];
        }
    } else {
#pragma unroll
        for (int r = 0; r < 8; ++r) { c0[r] = 0.0f; c1[r] = 0.0f; }
    }

    const float* __restrict__ ar0 = A + (size_t)(mBase + lm) * OUT_MF;       // A rows, tile 0
    const float* __restrict__ ar1 = A + (size_t)(mBase + 16 + lm) * OUT_MF;  // A rows, tile 1

    for (int k0 = 0; k0 < OUT_MF; k0 += 32) {
        if (k0 + 32 < OUT_MF) {                       // prefetch next k-tile
            __builtin_prefetch(ar0 + k0 + 32, 0, 1);  // -> global_prefetch_b8
            __builtin_prefetch(ar1 + k0 + 32, 0, 1);
        }
        bf16x16 a0, a1, b;
#pragma unroll
        for (int v = 0; v < 8; ++v) {
            int ka = (v < 4) ? (k0 + half * 8 + 2 * v)
                             : (k0 + 16 + half * 8 + 2 * (v - 4));
            float2 va0 = *(const float2*)(ar0 + ka);  // 8B-aligned (ka even)
            float2 va1 = *(const float2*)(ar1 + ka);
            a0[2 * v]     = f2bf(va0.x);
            a0[2 * v + 1] = f2bf(va0.y);
            a1[2 * v]     = f2bf(va1.x);
            a1[2 * v + 1] = f2bf(va1.y);
            int kb = ka;
            b[2 * v]      = f2bf(B[(size_t)kb * CL1_F + n]);
            b[2 * v + 1]  = f2bf(B[(size_t)(kb + 1) * CL1_F + n]);
        }
        // (neg_a, A, neg_b, B, c_mod, C, reuse_a, reuse_b)
        c0 = __builtin_amdgcn_wmma_f32_16x16x32_bf16(false, a0, false, b,
                                                     (short)0, c0, false, false);
        c1 = __builtin_amdgcn_wmma_f32_16x16x32_bf16(false, a1, false, b,
                                                     (short)0, c1, false, false);
    }
#pragma unroll
    for (int r = 0; r < 8; ++r) {
        C[(size_t)(crow0 + r) * CL1_F + n] = c0[r];
        C[(size_t)(crow1 + r) * CL1_F + n] = c1[r];
    }
}

// ---------------------------------------------------------------------------
// Head: bias + ReLU + BN(eval) + 64->2 linear
// ---------------------------------------------------------------------------
__global__ void k_head(const float* __restrict__ C64, const float* __restrict__ cheb_b,
                       const float* __restrict__ g, const float* __restrict__ bb,
                       const float* __restrict__ mean, const float* __restrict__ var,
                       const float* __restrict__ lin_w, const float* __restrict__ lin_b,
                       float* __restrict__ out)
{
    int node = blockIdx.x * blockDim.x + threadIdx.x;
    if (node >= N_NODES) return;
    const float* r = C64 + (size_t)node * CL1_F;
    float a0 = 0.0f, a1 = 0.0f;
#pragma unroll 8
    for (int i = 0; i < CL1_F; ++i) {
        float t = r[i] + cheb_b[i];
        t = fmaxf(t, 0.0f);
        t = (t - mean[i]) * (g[i] * rsqrtf(var[i] + EPSF)) + bb[i];
        a0 += t * lin_w[i];
        a1 += t * lin_w[CL1_F + i];
    }
    out[node * 2 + 0] = a0 + lin_b[0];
    out[node * 2 + 1] = a1 + lin_b[1];
}

// ---------------------------------------------------------------------------
extern "C" void kernel_launch(void* const* d_in, const int* in_sizes, int n_in,
                              void* d_out, int out_size, void* d_ws, size_t ws_size,
                              hipStream_t stream) {
    const float* x        = (const float*)d_in[0];
    const int*   ei       = (const int*)  d_in[1];
    const float* ew       = (const float*)d_in[2];
    const int*   smask    = (const int*)  d_in[3];
    const float* w_sp     = (const float*)d_in[4];
    const float* b_sp     = (const float*)d_in[5];
    const float* cheb_w   = (const float*)d_in[6]; // (3,1024,64)
    const float* cheb_b   = (const float*)d_in[7];
    const float* lin_w    = (const float*)d_in[8];
    const float* lin_b    = (const float*)d_in[9];
    const float* bns_g    = (const float*)d_in[10];
    const float* bns_b    = (const float*)d_in[11];
    const float* bns_m    = (const float*)d_in[12];
    const float* bns_v    = (const float*)d_in[13];
    const float* bn1_g    = (const float*)d_in[14];
    const float* bn1_b    = (const float*)d_in[15];
    const float* bn1_m    = (const float*)d_in[16];
    const float* bn1_v    = (const float*)d_in[17];
    float* out = (float*)d_out;

    const size_t NH = (size_t)N_NODES * OUT_MF; // 8M floats per feature map
    float* h   = (float*)d_ws;          // tx0
    float* t1  = h   + NH;              // tx1
    float* t2  = t1  + NH;              // lhat(tx1) -> tx2
    float* deg = t2  + NH;              // N
    float* nw  = deg + N_NODES;         // E
    float* c64 = nw  + E_EDGES;         // N x 64 accumulator

    // 1. sparse masked linear + relu + bn  -> h (tx0)
    k_sparse_linear<<<N_NODES, 256, 0, stream>>>(x, smask, w_sp, b_sp,
                                                 bns_g, bns_b, bns_m, bns_v, h);
    // 2. degree + edge normalization
    k_zero<<<(N_NODES + 255) / 256, 256, 0, stream>>>(deg, (size_t)N_NODES);
    k_deg<<<E_EDGES / 256, 256, 0, stream>>>(ei, ew, deg);
    k_normw<<<E_EDGES / 256, 256, 0, stream>>>(ei, ew, deg, nw);

    // 3. out = tx0 @ W0
    k_gemm_wmma<<<N_NODES / 64, 256, 0, stream>>>(h, cheb_w, c64, 0);

    // 4. tx1 = lhat(tx0); out += tx1 @ W1
    k_zero<<<(int)((NH + 255) / 256), 256, 0, stream>>>(t1, NH);
    k_spmm<<<E_EDGES, 256, 0, stream>>>(ei, nw, h, t1);
    k_gemm_wmma<<<N_NODES / 64, 256, 0, stream>>>(t1, cheb_w + (size_t)OUT_MF * CL1_F, c64, 1);

    // 5. tx2 = 2*lhat(tx1) - tx0; out += tx2 @ W2
    k_zero<<<(int)((NH + 255) / 256), 256, 0, stream>>>(t2, NH);
    k_spmm<<<E_EDGES, 256, 0, stream>>>(ei, nw, t1, t2);
    k_cheb_combine<<<(int)((NH + 255) / 256), 256, 0, stream>>>(t2, h, NH);
    k_gemm_wmma<<<N_NODES / 64, 256, 0, stream>>>(t2, cheb_w + 2 * (size_t)OUT_MF * CL1_F, c64, 1);

    // 6. bias + relu + bn1 + final linear
    k_head<<<N_NODES / 256, 256, 0, stream>>>(c64, cheb_b, bn1_g, bn1_b, bn1_m, bn1_v,
                                              lin_w, lin_b, out);
}
